// Network_6631429505511
// MI455X (gfx1250) — compile-verified, matrix operand order checked by reading, not run on hardware
//
#include <hip/hip_runtime.h>
#include <hip/hip_bf16.h>

// ---------------------------------------------------------------------------
// Problem constants (match reference)
// ---------------------------------------------------------------------------
#define NENT  100000   // multiple of 32 -> no M guards in GEMM
#define DFEA  200
#define DPAD  224      // DFEA padded to multiple of 32 (K pad)
#define D0FEA 100
#define D0PAD 128      // D0FEA padded to multiple of 32
#define NRELS 475
#define NBASE 100
#define NLAY  2
#define BNEPS 1e-5f
#define NENT_PAD 100032  // NENT padded to multiple of 64 (B rows of score GEMM)
#define WPADROWS 256     // weight-B rows padded to multiple of 64 (covers N=200)

typedef __attribute__((ext_vector_type(16))) _Float16 v16h;
typedef __attribute__((ext_vector_type(8)))  float    v8f;

union Frag { v16h h; uint4 q[2]; };

// ---------------------------------------------------------------------------
// CDNA5 async global->LDS copy (ASYNCcnt-tracked), bypasses VGPRs.
//   LDS byte address in vdst VGPR; 64-bit global address in vaddr
// ---------------------------------------------------------------------------
__device__ __forceinline__ void async_ld_b128(unsigned lds_off, const _Float16* g) {
    asm volatile("global_load_async_to_lds_b128 %0, %1, off"
                 :: "v"(lds_off), "v"(g) : "memory");
}
__device__ __forceinline__ void wait_async0() {
    asm volatile("s_wait_asynccnt 0x0" ::: "memory");
}

// ---------------------------------------------------------------------------
// WMMA f16 GEMM (guard-free loads, double-buffered async LDS staging):
//   C[M,Nn](f32) = A[M,Kpad](f16) @ B_t + bias
//   B_t is [Npad, Kpad] row-major (B transposed), rows/K zero-padded.
//   M % 32 == 0. Block tile 32x64 (2x4 waves of 16x16). K step 32.
// ---------------------------------------------------------------------------
#define TM 32
#define TN 64
#define TK 32

__global__ __launch_bounds__(256)
void gemm_wmma_f16(const _Float16* __restrict__ A, const _Float16* __restrict__ Bt,
                   const float* __restrict__ bias, float* __restrict__ C,
                   int Nn, int Kpad, int lda, int ldb, int ldc) {
    __shared__ __align__(16) _Float16 As[2][TM][TK];  // 2 x 2KB
    __shared__ __align__(16) _Float16 Bs[2][TN][TK];  // 2 x 4KB

    const int tid  = threadIdx.x;
    const int lane = tid & 31;
    const int wid  = tid >> 5;
    const int wrow = wid >> 2;   // 0..1
    const int wcol = wid & 3;    // 0..3
    const int m0 = blockIdx.y * TM;
    const int n0 = blockIdx.x * TN;

    // staging assignment: B = 256 b128 chunks (all threads), A = 128 (tid<128)
    const int  br  = tid >> 2, bch = tid & 3;
    const int  ar  = (tid & 127) >> 2, ach = tid & 3;
    const bool doA = tid < 128;

    const _Float16* gB = Bt + (long)(n0 + br) * ldb + bch * 8;
    const _Float16* gA = A  + (long)(m0 + ar) * lda + ach * 8;

    unsigned ldsB[2], ldsA[2];
    ldsB[0] = (unsigned)(unsigned long long)&Bs[0][br][bch * 8];
    ldsB[1] = (unsigned)(unsigned long long)&Bs[1][br][bch * 8];
    ldsA[0] = (unsigned)(unsigned long long)&As[0][ar][ach * 8];
    ldsA[1] = (unsigned)(unsigned long long)&As[1][ar][ach * 8];

    // prologue: tile 0 -> buffer 0
    async_ld_b128(ldsB[0], gB);
    if (doA) async_ld_b128(ldsA[0], gA);

    const int mrow = wrow * 16 + (lane & 15);
    const int ncol = wcol * 16 + (lane & 15);
    const int hi4  = (lane >= 16) ? 1 : 0;

    v8f acc = {};
    const int ntiles = Kpad / TK;
    for (int t = 0; t < ntiles; ++t) {
        wait_async0();        // my copies of tile t have landed
        __syncthreads();      // everyone's have; all reads of other buffer done
        if (t + 1 < ntiles) { // prefetch tile t+1 into the other buffer
            int nb = (t + 1) & 1;
            async_ld_b128(ldsB[nb], gB + (t + 1) * TK);
            if (doA) async_ld_b128(ldsA[nb], gA + (t + 1) * TK);
        }
        // Fragments: per CDNA5 16-bit layout each lane = two contiguous 16B
        // chunks of one LDS row -> 2x ds_load_b128 per fragment.
        const int cur = t & 1;
        Frag fa, fb;
        const uint4* arow = (const uint4*)&As[cur][mrow][0];
        const uint4* brow = (const uint4*)&Bs[cur][ncol][0];
        fa.q[0] = arow[hi4];     fa.q[1] = arow[2 + hi4];
        fb.q[0] = brow[hi4];     fb.q[1] = brow[2 + hi4];

        acc = __builtin_amdgcn_wmma_f32_16x16x32_f16(
                  false, fa.h, false, fb.h, (short)0, acc, false, false);
    }

    // C/D layout: lane col = lane&15; VGPR r -> m = r + 8*(lane>=16)
    const int nn = n0 + ncol;
    const float bv = (bias != nullptr && nn < Nn) ? bias[nn] : 0.f;
    const int mbase = m0 + wrow * 16 + ((lane >= 16) ? 8 : 0);
    #pragma unroll
    for (int r = 0; r < 8; ++r) {
        if (nn < Nn) C[(long)(mbase + r) * ldc + nn] = acc[r] + bv;
    }
}

// ---------------------------------------------------------------------------
// f32 -> f16 with K padding: out[r*ldo + c], zeros for c >= cols
// ---------------------------------------------------------------------------
__global__ void cvt_pad_f16(const float* __restrict__ in, _Float16* __restrict__ out,
                            long rows, int cols, int ldi, int ldo) {
    long i = (long)blockIdx.x * blockDim.x + threadIdx.x;
    if (i >= rows * (long)ldo) return;
    long r = i / ldo; int c = (int)(i % ldo);
    out[i] = (c < cols) ? (_Float16)in[r * ldi + c] : (_Float16)0.f;
}

// ---------------------------------------------------------------------------
// f32 [K,N] -> f16 transposed+padded [Npad, Kpad]
// ---------------------------------------------------------------------------
__global__ void cvt_transpose_pad_f16(const float* __restrict__ in, _Float16* __restrict__ out,
                                      int K, int N, int npad, int kpad) {
    long i = (long)blockIdx.x * blockDim.x + threadIdx.x;
    if (i >= (long)npad * kpad) return;
    int n = (int)(i / kpad), k = (int)(i % kpad);
    out[i] = (n < N && k < K) ? (_Float16)in[(long)k * N + n] : (_Float16)0.f;
}

// ---------------------------------------------------------------------------
// Small scalar GEMM for rel matrices (<= 475x200x200)
// ---------------------------------------------------------------------------
__global__ void gemm_small(const float* __restrict__ A, const float* __restrict__ B,
                           float* __restrict__ C, int M, int N, int K) {
    int idx = blockIdx.x * blockDim.x + threadIdx.x;
    if (idx >= M * N) return;
    int m = idx / N, n = idx % N;
    float s = 0.f;
    for (int k = 0; k < K; ++k) s += A[m * K + k] * B[k * N + n];
    C[idx] = s;
}

// ---------------------------------------------------------------------------
// Edge gather * rel, scatter-add (float4 lanes, atomics resolve in 192MB L2)
// ---------------------------------------------------------------------------
__global__ void edge_scatter(const int* __restrict__ src_id, const int* __restrict__ dst_id,
                             const int* __restrict__ e_type,
                             const float* __restrict__ all_ent,
                             const float* __restrict__ rel_embed,
                             float* __restrict__ agg, int E_, int N_, int self_rel) {
    long idx = (long)blockIdx.x * blockDim.x + threadIdx.x;
    long total = (long)(E_ + N_) * (DFEA / 4);
    if (idx >= total) return;
    int e = (int)(idx / (DFEA / 4));
    int c = (int)(idx % (DFEA / 4)) * 4;
    int src, dst, et;
    if (e < E_) { src = src_id[e]; dst = dst_id[e]; et = e_type[e]; }
    else        { src = e - E_;   dst = src;        et = self_rel;  }
    const float4 s = *(const float4*)(all_ent   + (long)src * DFEA + c);
    const float4 r = *(const float4*)(rel_embed + (long)et  * DFEA + c);
    float* o = agg + (long)dst * DFEA + c;
    atomicAdd(o + 0, s.x * r.x);
    atomicAdd(o + 1, s.y * r.y);
    atomicAdd(o + 2, s.z * r.z);
    atomicAdd(o + 3, s.w * r.w);
}

// ---------------------------------------------------------------------------
// BatchNorm stats: per-feature sum / sumsq (block partials + global atomics)
// ---------------------------------------------------------------------------
#define BN_ROWS 256
__global__ void bn_stats(const float* __restrict__ x, float* __restrict__ sum,
                         float* __restrict__ sumsq, int rows, int cols) {
    int f = threadIdx.x;
    if (f >= cols) return;
    long r0 = (long)blockIdx.x * BN_ROWS;
    long r1 = r0 + BN_ROWS; if (r1 > rows) r1 = rows;
    float s = 0.f, ss = 0.f;
    for (long r = r0; r < r1; ++r) {
        float v = x[r * cols + f];
        s += v; ss += v * v;
    }
    atomicAdd(&sum[f], s);
    atomicAdd(&sumsq[f], ss);
}

// ---------------------------------------------------------------------------
// Fused BN(batch stats) + ReLU
//   out16: padded stride ld16 (zeros in padding), out32 (optional): stride cols
// ---------------------------------------------------------------------------
__global__ void bn_relu(const float* __restrict__ x,
                        const float* __restrict__ sum, const float* __restrict__ sumsq,
                        const float* __restrict__ g, const float* __restrict__ b,
                        _Float16* __restrict__ out16, float* __restrict__ out32,
                        long rows, int cols, int ld16, float invN) {
    long i = (long)blockIdx.x * blockDim.x + threadIdx.x;
    if (i >= rows * (long)ld16) return;
    long r = i / ld16; int f = (int)(i % ld16);
    if (f >= cols) { out16[i] = (_Float16)0.f; return; }
    float m = sum[f] * invN;
    float v = sumsq[f] * invN - m * m;          // biased variance (jnp.var)
    float y = (x[r * cols + f] - m) * rsqrtf(v + BNEPS) * g[f] + b[f];
    y = fmaxf(y, 0.f);
    out16[i] = (_Float16)y;
    if (out32) out32[r * cols + f] = y;
}

// ---------------------------------------------------------------------------
// q[b,d] = all_ent[subj[b],d] * rel_embed[rel[b],d] -> f16, padded stride
// ---------------------------------------------------------------------------
__global__ void make_q(const int* __restrict__ subj, const int* __restrict__ rel,
                       const float* __restrict__ all_ent, const float* __restrict__ rel_embed,
                       _Float16* __restrict__ q, int Bq) {
    int idx = blockIdx.x * blockDim.x + threadIdx.x;
    if (idx >= Bq * DPAD) return;
    int b = idx / DPAD, d = idx % DPAD;
    q[idx] = (d < DFEA)
        ? (_Float16)(all_ent[(long)subj[b] * DFEA + d] * rel_embed[(long)rel[b] * DFEA + d])
        : (_Float16)0.f;
}

// ---------------------------------------------------------------------------
// Zero f16 rows (for ent16 row padding 100000..100031)
// ---------------------------------------------------------------------------
__global__ void zero_f16(_Float16* __restrict__ p, long n) {
    long i = (long)blockIdx.x * blockDim.x + threadIdx.x;
    if (i < n) p[i] = (_Float16)0.f;
}

// ---------------------------------------------------------------------------
// Host orchestration
// ---------------------------------------------------------------------------
static inline size_t alignUp(size_t x) { return (x + 255) & ~(size_t)255; }

extern "C" void kernel_launch(void* const* d_in, const int* in_sizes, int n_in,
                              void* d_out, int out_size, void* d_ws, size_t ws_size,
                              hipStream_t stream) {
    const int* src_id = (const int*)d_in[0];
    const int* dst_id = (const int*)d_in[1];
    const int* e_type = (const int*)d_in[2];
    const int* subj   = (const int*)d_in[3];
    const int* rel    = (const int*)d_in[4];
    const float* emb_h    = (const float*)d_in[5];
    const float* emb_e    = (const float*)d_in[6];
    const float* lin_w    = (const float*)d_in[7];
    const float* lin_b    = (const float*)d_in[8];
    const float* rel_wt   = (const float*)d_in[9];
    const float* w_rel    = (const float*)d_in[10];
    const float* agg_bn_g = (const float*)d_in[11];
    const float* agg_bn_b = (const float*)d_in[12];
    const float* cell_bn_g = (const float*)d_in[13];
    const float* cell_bn_b = (const float*)d_in[14];
    const float* concat_w  = (const float*)d_in[15];
    const float* concat_b  = (const float*)d_in[16];
    float* score = (float*)d_out;

    const int E  = in_sizes[0];
    const int Bq = in_sizes[3];   // 1024, multiple of 32

    // ---- workspace layout ----
    char* w = (char*)d_ws;
    float* all_ent = (float*)w;                 w += alignUp((size_t)NENT * DFEA * 4);
    float* bufB    = (float*)w;                 w += alignUp((size_t)NENT * DFEA * 4); // agg / h
    _Float16* ent16  = (_Float16*)w;            w += alignUp((size_t)NENT_PAD * DPAD * 2);
    _Float16* aggn16 = (_Float16*)w;            w += alignUp((size_t)NENT * DPAD * 2);
    _Float16* embh16 = (_Float16*)w;            w += alignUp((size_t)NENT * D0PAD * 2);
    _Float16* linw16 = (_Float16*)w;            w += alignUp((size_t)WPADROWS * D0PAD * 2);
    _Float16* cw16   = (_Float16*)w;            w += alignUp((size_t)NLAY * WPADROWS * DPAD * 2);
    float* relA = (float*)w;                    w += alignUp((size_t)NRELS * DFEA * 4);
    float* relB = (float*)w;                    w += alignUp((size_t)NRELS * DFEA * 4);
    _Float16* q16 = (_Float16*)w;               w += alignUp((size_t)Bq * DPAD * 2);
    float* bn_sum   = (float*)w;                w += alignUp((size_t)DFEA * 4);
    float* bn_sumsq = (float*)w;                w += alignUp((size_t)DFEA * 4);

    const float invN = 1.0f / (float)NENT;
    auto blk = [](long n, int t) { return (unsigned)((n + t - 1) / t); };

    // ---- f16 operand prep (padded / transposed layouts) ----
    cvt_pad_f16<<<blk((long)NENT * D0PAD, 256), 256, 0, stream>>>(
        emb_h, embh16, NENT, D0FEA, D0FEA, D0PAD);
    cvt_transpose_pad_f16<<<blk((long)WPADROWS * D0PAD, 256), 256, 0, stream>>>(
        lin_w, linw16, D0FEA, DFEA, WPADROWS, D0PAD);
    for (int i = 0; i < NLAY; ++i)
        cvt_transpose_pad_f16<<<blk((long)WPADROWS * DPAD, 256), 256, 0, stream>>>(
            concat_w + (size_t)i * DFEA * DFEA, cw16 + (size_t)i * WPADROWS * DPAD,
            DFEA, DFEA, WPADROWS, DPAD);
    zero_f16<<<blk((long)(NENT_PAD - NENT) * DPAD, 256), 256, 0, stream>>>(
        ent16 + (size_t)NENT * DPAD, (long)(NENT_PAD - NENT) * DPAD);

    // rel_embed = rel_wt @ emb_e
    gemm_small<<<blk((long)NRELS * DFEA, 256), 256, 0, stream>>>(
        rel_wt, emb_e, relA, NRELS, DFEA, NBASE);

    // all_ent = emb_h @ lin_w + lin_b   (WMMA)
    {
        dim3 g(blk(DFEA, TN), NENT / TM);
        gemm_wmma_f16<<<g, 256, 0, stream>>>(embh16, linw16, lin_b, all_ent,
                                             DFEA, D0PAD, D0PAD, D0PAD, DFEA);
    }

    // ---- layers ----
    for (int i = 0; i < NLAY; ++i) {
        float* rel_cur  = (i == 0) ? relA : relB;
        float* rel_next = (i == 0) ? relB : relA;

        hipMemsetAsync(bufB, 0, (size_t)NENT * DFEA * 4, stream);
        edge_scatter<<<blk((long)(E + NENT) * (DFEA / 4), 256), 256, 0, stream>>>(
            src_id, dst_id, e_type, all_ent, rel_cur, bufB, E, NENT, NRELS - 1);

        hipMemsetAsync(bn_sum, 0, DFEA * 4, stream);
        hipMemsetAsync(bn_sumsq, 0, DFEA * 4, stream);
        bn_stats<<<blk(NENT, BN_ROWS), 256, 0, stream>>>(bufB, bn_sum, bn_sumsq, NENT, DFEA);
        bn_relu<<<blk((long)NENT * DPAD, 256), 256, 0, stream>>>(
            bufB, bn_sum, bn_sumsq, agg_bn_g + i * DFEA, agg_bn_b + i * DFEA,
            aggn16, nullptr, NENT, DFEA, DPAD, invN);

        // h = agg_norm @ concat_w[i] + concat_b[i]  (WMMA); overwrites bufB
        {
            dim3 g(blk(DFEA, TN), NENT / TM);
            gemm_wmma_f16<<<g, 256, 0, stream>>>(
                aggn16, cw16 + (size_t)i * WPADROWS * DPAD,
                concat_b + i * DFEA, bufB, DFEA, DPAD, DPAD, DPAD, DFEA);
        }

        hipMemsetAsync(bn_sum, 0, DFEA * 4, stream);
        hipMemsetAsync(bn_sumsq, 0, DFEA * 4, stream);
        bn_stats<<<blk(NENT, BN_ROWS), 256, 0, stream>>>(bufB, bn_sum, bn_sumsq, NENT, DFEA);
        bn_relu<<<blk((long)NENT * DPAD, 256), 256, 0, stream>>>(
            bufB, bn_sum, bn_sumsq, cell_bn_g + i * DFEA, cell_bn_b + i * DFEA,
            ent16, all_ent, NENT, DFEA, DPAD, invN);

        // rel_embed = rel_embed @ w_rel
        gemm_small<<<blk((long)NRELS * DFEA, 256), 256, 0, stream>>>(
            rel_cur, w_rel, rel_next, NRELS, DFEA, DFEA);
    }

    // ---- score = (all_ent[subj] * rel_final) @ all_ent.T  (WMMA, Bt = ent16) ----
    float* rel_final = relA;  // after 2 ping-pong updates
    make_q<<<blk((long)Bq * DPAD, 256), 256, 0, stream>>>(subj, rel, all_ent, rel_final, q16, Bq);
    {
        dim3 g(NENT_PAD / TN, Bq / TM);
        gemm_wmma_f16<<<g, 256, 0, stream>>>(q16, ent16, nullptr, score,
                                             NENT, DPAD, DPAD, DPAD, NENT);
    }
}
